// proteusAI_16630113370597
// MI455X (gfx1250) — matrix operand order repaired and checked by reading, not compiled, not run on hardware
//
#include <hip/hip_runtime.h>
#include <hip/hip_bf16.h>

typedef __attribute__((ext_vector_type(16))) _Float16 v16h;
typedef __attribute__((ext_vector_type(8)))  _Float16 h8;
typedef __attribute__((ext_vector_type(4)))  _Float16 h4;
typedef __attribute__((ext_vector_type(8)))  float    v8f;
typedef __attribute__((ext_vector_type(4)))  unsigned int u32x4;
typedef __attribute__((ext_vector_type(8)))  int      i32x8;
typedef __attribute__((ext_vector_type(4)))  int      i32x4;

#define B_   2
#define N_   2048
#define D_   512
#define H_   16
#define DK_  32
#define SCALE_ 0.17677669529663687f   // 1/sqrt(32)
#define NEG_ (-1.0e9f)

#if defined(__gfx1250__) && __has_builtin(__builtin_amdgcn_tensor_load_to_lds) && \
    __has_builtin(__builtin_amdgcn_s_wait_tensorcnt)
#define USE_TDM 1
#endif

__device__ __forceinline__ v16h pack16(h8 lo, h8 hh) {
  v16h a;
#pragma unroll
  for (int t = 0; t < 8; ++t) { a[t] = lo[t]; a[t + 8] = hh[t]; }
  return a;
}
// A fragment (16x32 f16): lane's 16 contiguous halves at base + hi*16
__device__ __forceinline__ v16h load_a(const _Float16* base, int hi) {
  return pack16(*(const h8*)(base + hi * 16), *(const h8*)(base + hi * 16 + 8));
}
// B fragment: 16 contiguous halves at base
__device__ __forceinline__ v16h load_b16(const _Float16* base) {
  return pack16(*(const h8*)(base), *(const h8*)(base + 8));
}

__device__ __forceinline__ v8f wmma16(v16h a, v16h b, v8f c) {
  return __builtin_amdgcn_wmma_f32_16x16x32_f16(false, a, false, b, (short)0, c,
                                                false, false);
}

__device__ __forceinline__ float rmax16(float v) {
#pragma unroll
  for (int m = 1; m < 16; m <<= 1) v = fmaxf(v, __shfl_xor(v, m, 32));
  return v;
}
__device__ __forceinline__ float rsum16(float v) {
#pragma unroll
  for (int m = 1; m < 16; m <<= 1) v += __shfl_xor(v, m, 32);
  return v;
}

// ---------------------------------------------------------------------------
// Kernel A: f32 -> f16 convert of q,k,v activations (vectorized, coalesced)
// ---------------------------------------------------------------------------
__global__ void __launch_bounds__(256) k_cvt(
    const float* __restrict__ a, const float* __restrict__ b,
    const float* __restrict__ c, _Float16* __restrict__ oa,
    _Float16* __restrict__ ob, _Float16* __restrict__ oc) {
  size_t i = ((size_t)blockIdx.x * blockDim.x + threadIdx.x) * 4;
  if (i + 3 < (size_t)B_ * N_ * D_) {
    float4 fa = *(const float4*)(a + i);
    float4 fb = *(const float4*)(b + i);
    float4 fc = *(const float4*)(c + i);
    *(h4*)(oa + i) = (h4){(_Float16)fa.x, (_Float16)fa.y, (_Float16)fa.z, (_Float16)fa.w};
    *(h4*)(ob + i) = (h4){(_Float16)fb.x, (_Float16)fb.y, (_Float16)fb.z, (_Float16)fb.w};
    *(h4*)(oc + i) = (h4){(_Float16)fc.x, (_Float16)fc.y, (_Float16)fc.z, (_Float16)fc.w};
  }
}

// ---------------------------------------------------------------------------
// Kernel B: weight prep. Pt[h][dk][d] = (f16) proj[h][d][dk] (x3),
//           Wh[o][d]    = (f16) w_out[o][d]
// ---------------------------------------------------------------------------
__global__ void __launch_bounds__(256) k_prep(
    const float* __restrict__ qp, const float* __restrict__ kp,
    const float* __restrict__ vp, const float* __restrict__ w,
    _Float16* __restrict__ Ptq, _Float16* __restrict__ Ptk,
    _Float16* __restrict__ Ptv, _Float16* __restrict__ Wh) {
  int i = blockIdx.x * blockDim.x + threadIdx.x;
  const int np = H_ * D_ * DK_;
  if (i < np) {
    int h = i / (D_ * DK_);
    int r = i % (D_ * DK_);
    int d = r / DK_, dk = r % DK_;
    size_t o = ((size_t)h * DK_ + dk) * D_ + d;
    Ptq[o] = (_Float16)qp[i];
    Ptk[o] = (_Float16)kp[i];
    Ptv[o] = (_Float16)vp[i];
  } else if (i < np + D_ * D_) {
    int j = i - np;
    Wh[j] = (_Float16)w[j];
  }
}

// ---------------------------------------------------------------------------
// Kernel 1: per-head QKV projections.  Q,K -> f16 [B,H,N,DK];
// V -> f16 [B,H,DK,N] (pre-transposed so k_attn's P x V B-fragments and the
// 2-D TDM tile fetch are contiguous). B-fragments double-buffered across heads.
// ---------------------------------------------------------------------------
__global__ void __launch_bounds__(128) k_proj(
    const _Float16* __restrict__ Xq, const _Float16* __restrict__ Xk,
    const _Float16* __restrict__ Xv, const _Float16* __restrict__ Ptq,
    const _Float16* __restrict__ Ptk, const _Float16* __restrict__ Ptv,
    const float* __restrict__ qb, const float* __restrict__ kb,
    const float* __restrict__ vb, _Float16* __restrict__ Qh,
    _Float16* __restrict__ Kh, _Float16* __restrict__ Vh) {
  const int wid = threadIdx.x >> 5, lane = threadIdx.x & 31;
  const int hi = lane >> 4, col = lane & 15;
  const int wtid = blockIdx.x * 4 + wid;

  const int per = B_ * (N_ / 16) * (H_ / 4);   // 1024 per input tensor
  int which = wtid / per;
  int r = wtid % per;
  int b = r / ((N_ / 16) * (H_ / 4));
  r %= (N_ / 16) * (H_ / 4);
  int ntile = r / (H_ / 4);
  int h0 = (r % (H_ / 4)) * 4;

  const _Float16* X  = which == 0 ? Xq  : (which == 1 ? Xk  : Xv);
  const _Float16* Pt = which == 0 ? Ptq : (which == 1 ? Ptk : Ptv);
  const float*    Bb = which == 0 ? qb  : (which == 1 ? kb  : vb);
  _Float16*       O  = which == 0 ? Qh  : (which == 1 ? Kh  : Vh);

  const int arow = ntile * 16 + col;           // A-matrix row (m = lane&15)
  const _Float16* Xr = X + ((size_t)b * N_ + arow) * D_;

  v8f acc[8];
#pragma unroll
  for (int i = 0; i < 8; ++i) acc[i] = (v8f){};

  for (int kc = 0; kc < D_; kc += 32) {
    v16h a = load_a(Xr + kc, hi);
    // one-head lookahead so B loads overlap the previous head's WMMAs
    const _Float16* Pb = Pt + ((size_t)h0 * DK_ + col) * D_ + kc + hi * 16;
    v16h nb0 = load_b16(Pb);
    v16h nb1 = load_b16(Pb + (size_t)16 * D_);
#pragma unroll
    for (int hh = 0; hh < 4; ++hh) {
      v16h b0 = nb0, b1 = nb1;
      if (hh < 3) {
        const _Float16* Pn = Pt + ((size_t)(h0 + hh + 1) * DK_ + col) * D_ + kc + hi * 16;
        nb0 = load_b16(Pn);
        nb1 = load_b16(Pn + (size_t)16 * D_);
      }
      acc[hh * 2]     = wmma16(a, b0, acc[hh * 2]);
      acc[hh * 2 + 1] = wmma16(a, b1, acc[hh * 2 + 1]);
    }
  }
#pragma unroll
  for (int hh = 0; hh < 4; ++hh) {
    int h = h0 + hh;
#pragma unroll
    for (int g = 0; g < 8; ++g) {
      int rr = g + 8 * hi;                      // C-matrix row (local)
      _Float16 v0 = (_Float16)(acc[hh * 2][g]     + Bb[h * DK_ + col]);
      _Float16 v1 = (_Float16)(acc[hh * 2 + 1][g] + Bb[h * DK_ + 16 + col]);
      if (which == 2) {  // V: transposed layout [B,H,DK,N]
        _Float16* Oh = O + (((size_t)b * H_ + h) * DK_) * N_ + ntile * 16 + rr;
        Oh[(size_t)col * N_]        = v0;
        Oh[(size_t)(16 + col) * N_] = v1;
      } else {           // Q,K: [B,H,N,DK]
        _Float16* Oh = O + (((size_t)b * H_ + h) * N_ + ntile * 16 + rr) * DK_;
        Oh[col]      = v0;
        Oh[16 + col] = v1;
      }
    }
  }
}

// ---------------------------------------------------------------------------
// Kernel 2: fused RBF-masked flash attention.
// Block = 8 waves = 128 query rows of one (b,h); j swept in 32-row tiles.
// K tile: 1-D TDM descriptor.  V tile: 2-D TDM descriptor straight from the
// transposed global V into [dk][j] LDS.  Fallback: coalesced copies.
// ---------------------------------------------------------------------------
__global__ void __launch_bounds__(256) k_attn(
    const _Float16* __restrict__ Qh, const _Float16* __restrict__ Kh,
    const _Float16* __restrict__ Vh, const float* __restrict__ coords,
    const float* __restrict__ spreads, _Float16* __restrict__ Af) {
  __shared__ _Float16 Kt[32 * DK_];       // [j][dk]
  __shared__ _Float16 VtT[DK_ * 32];      // [dk][j]
  __shared__ float    Cj[32 * 3];
  __shared__ _Float16 Pld[8][16 * 32];
  __shared__ float    Qc[8][16 * 3];

  const int wid = threadIdx.x >> 5, lane = threadIdx.x & 31;
  const int hi = lane >> 4, col = lane & 15;

  int bid = blockIdx.x;
  int b = bid / (H_ * (N_ / 128));
  int r = bid % (H_ * (N_ / 128));
  int h = r / (N_ / 128);
  int itile = r % (N_ / 128);
  const int i0 = itile * 128 + wid * 16;

  const size_t bh = (size_t)b * H_ + h;
  const _Float16* Qp = Qh + (bh * N_ + i0) * DK_;
  const _Float16* Kp = Kh + bh * N_ * DK_;
  const _Float16* Vp = Vh + bh * DK_ * N_;   // transposed: [dk][n]

#ifdef USE_TDM
  const unsigned int ldsK = (unsigned int)(size_t)(void*)Kt;
  const unsigned int ldsV = (unsigned int)(size_t)(void*)VtT;
#endif

  // rbf^2 = exp(-d2/s^2); band [0.01,0.99] on rbf <=> d2 in [dlo, dhi]
  float s = spreads[h];
  float s2 = s * s;
  float inv_s2 = 1.0f / s2;
  float dlo = 0.0201006717f * s2;   // -2 ln(0.99) * s2
  float dhi = 9.2103403720f * s2;   // -2 ln(0.01) * s2

  v16h aq = load_a(Qp + (size_t)col * DK_, hi);

  if (lane < 16) {
#pragma unroll
    for (int c = 0; c < 3; ++c)
      Qc[wid][lane * 3 + c] = coords[((size_t)b * N_ + i0 + lane) * 3 + c];
  }

  float m_i[8], l_i[8];
#pragma unroll
  for (int g = 0; g < 8; ++g) { m_i[g] = -3.0e38f; l_i[g] = 0.0f; }
  v8f o0 = {}, o1 = {};

  for (int j0 = 0; j0 < N_; j0 += 32) {
    __syncthreads();   // previous tile fully consumed
#ifdef USE_TDM
    if (wid == 0) {
      unsigned long long gaK = (unsigned long long)(const void*)(Kp + (size_t)j0 * DK_);
      unsigned long long gaV = (unsigned long long)(const void*)(Vp + j0);
      // D# group0: count=1 | lds_addr | global_addr | type=2
      u32x4 g0K = (u32x4){1u, ldsK, (unsigned int)gaK,
                          ((unsigned int)(gaK >> 32) & 0x01FFFFFFu) | 0x80000000u};
      u32x4 g0V = (u32x4){1u, ldsV, (unsigned int)gaV,
                          ((unsigned int)(gaV >> 32) & 0x01FFFFFFu) | 0x80000000u};
      // K: 1-D, 1024 contiguous 2B elements
      i32x8 g1K = (i32x8){(int)(1u << 16),     // data_size = 1 (2 bytes)
                          (int)(1024u << 16),  // tensor_dim0 = 1024
                          (int)(1u << 16),     // tensor_dim1 = 1
                          (int)(1024u << 16),  // tile_dim0 = 1024
                          1,                   // tile_dim1 = 1
                          1024, 0, 0};         // tensor_dim0_stride = 1024
      // V: 2-D, 32x32 tile of rows with stride N_ (transposed V)
      i32x8 g1V = (i32x8){(int)(1u << 16),           // data_size = 1 (2 bytes)
                          (int)((unsigned)N_ << 16), // tensor_dim0 = 2048
                          (int)(32u << 16),          // tensor_dim1 = 32
                          (int)(32u << 16),          // tile_dim0 = 32
                          32,                        // tile_dim1 = 32
                          N_, 0, 0};                 // tensor_dim0_stride = 2048
      i32x4 gz = (i32x4){0, 0, 0, 0};
#if __clang_major__ >= 23
      i32x8 gz8 = (i32x8){0, 0, 0, 0, 0, 0, 0, 0};
      __builtin_amdgcn_tensor_load_to_lds(g0K, g1K, gz, gz, gz8, 0);
      __builtin_amdgcn_tensor_load_to_lds(g0V, g1V, gz, gz, gz8, 0);
#else
      __builtin_amdgcn_tensor_load_to_lds(g0K, g1K, gz, gz, 0);
      __builtin_amdgcn_tensor_load_to_lds(g0V, g1V, gz, gz, 0);
#endif
      __builtin_amdgcn_s_wait_tensorcnt(0);
    }
#else
    {
      int tn = threadIdx.x;
      if (tn < 128) {  // K: 2048 B contiguous, 128 x uint4
        ((uint4*)Kt)[tn] = ((const uint4*)(Kp + (size_t)j0 * DK_))[tn];
      } else {         // V: 16 dk rows x 32 halves from transposed V, 8 thr/row
        int t = tn - 128;
        int dk = t >> 3, off = (t & 7) * 4;
        *(h4*)(VtT + dk * 32 + off) = *(const h4*)(Vp + (size_t)dk * N_ + j0 + off);
      }
    }
    if (j0 + 32 < N_) {  // warm caches for the next tile (global_prefetch_b8)
      __builtin_prefetch(Kp + (size_t)(j0 + 32) * DK_ + lane * 32, 0, 1);
      __builtin_prefetch(Vp + (size_t)lane * N_ + j0 + 32, 0, 1);
    }
#endif
    if (threadIdx.x < 96)
      Cj[threadIdx.x] = coords[((size_t)b * N_ + j0) * 3 + threadIdx.x];
    __syncthreads();   // staged data visible

    // S = Q x K^T : B element (kdim=dk, n=jlocal) = Kt[jlocal][dk], contiguous
    v16h bk0 = load_b16(Kt + (size_t)col * DK_ + hi * 16);
    v16h bk1 = load_b16(Kt + (size_t)(16 + col) * DK_ + hi * 16);
    v8f zero = {};
    v8f s0 = wmma16(aq, bk0, zero);
    v8f s1 = wmma16(aq, bk1, zero);

    float kx0 = Cj[col * 3 + 0], ky0 = Cj[col * 3 + 1], kz0 = Cj[col * 3 + 2];
    float kx1 = Cj[(16 + col) * 3 + 0], ky1 = Cj[(16 + col) * 3 + 1],
          kz1 = Cj[(16 + col) * 3 + 2];

#pragma unroll
    for (int g = 0; g < 8; ++g) {
      int rl = g + 8 * hi;   // local C row
      float qx = Qc[wid][rl * 3 + 0], qy = Qc[wid][rl * 3 + 1],
            qz = Qc[wid][rl * 3 + 2];
      float dx = qx - kx0, dy = qy - ky0, dz = qz - kz0;
      float d2a = dx * dx + dy * dy + dz * dz;
      dx = qx - kx1; dy = qy - ky1; dz = qz - kz1;
      float d2b = dx * dx + dy * dy + dz * dz;
      bool va = (d2a >= dlo) && (d2a <= dhi);
      bool vb = (d2b >= dlo) && (d2b <= dhi);
      float sa = va ? s0[g] * SCALE_ * __expf(-d2a * inv_s2) : NEG_;
      float sb = vb ? s1[g] * SCALE_ * __expf(-d2b * inv_s2) : NEG_;

      float mr = rmax16(fmaxf(sa, sb));
      float mn = fmaxf(m_i[g], mr);
      float alpha = __expf(m_i[g] - mn);
      float pa = __expf(sa - mn);
      float pb = __expf(sb - mn);
      float ls = rsum16(pa + pb);
      m_i[g] = mn;
      l_i[g] = l_i[g] * alpha + ls;
      o0[g] *= alpha;
      o1[g] *= alpha;
      Pld[wid][rl * 32 + col]      = (_Float16)pa;
      Pld[wid][rl * 32 + 16 + col] = (_Float16)pb;
    }
    asm volatile("s_wait_dscnt 0" ::: "memory");  // P stores visible to re-swizzle loads

    // O += P x V : A = P (16x32) per-wave LDS, B from VtT (contiguous)
    v16h ap  = load_a(&Pld[wid][(size_t)col * 32], hi);
    v16h bv0 = load_b16(VtT + (size_t)col * 32 + hi * 16);
    v16h bv1 = load_b16(VtT + (size_t)(16 + col) * 32 + hi * 16);
    o0 = wmma16(ap, bv0, o0);
    o1 = wmma16(ap, bv1, o1);
  }

#pragma unroll
  for (int g = 0; g < 8; ++g) {
    float inv_l = 1.0f / l_i[g];
    int i = i0 + g + 8 * hi;
    _Float16* Ar = Af + ((size_t)b * N_ + i) * D_;
    Ar[col * H_ + h]        = (_Float16)(o0[g] * inv_l);   // d = dk*H + h
    Ar[(16 + col) * H_ + h] = (_Float16)(o1[g] * inv_l);
  }
}

// ---------------------------------------------------------------------------
// Kernel 3: out = Af @ w_out^T (A f16, B = f16 Wh [o][d], f32 out).
// Fragments double-buffered across the K loop.
// ---------------------------------------------------------------------------
__global__ void __launch_bounds__(128) k_outproj(const _Float16* __restrict__ Af,
                                                 const _Float16* __restrict__ Wh,
                                                 float* __restrict__ out) {
  const int wid = threadIdx.x >> 5, lane = threadIdx.x & 31;
  const int hi = lane >> 4, col = lane & 15;
  int wtid = blockIdx.x * 4 + wid;
  int mtile = wtid / (D_ / 32);
  int ctile = wtid % (D_ / 32);
  int arow = mtile * 16 + col;
  int c0 = ctile * 32;
  const _Float16* Ar = Af + (size_t)arow * D_;
  const _Float16* W0 = Wh + (size_t)(c0 + col) * D_;
  const _Float16* W1 = Wh + (size_t)(c0 + 16 + col) * D_;

  v8f acc0 = {}, acc1 = {};
  v16h a  = load_a(Ar, hi);
  v16h b0 = load_b16(W0 + hi * 16);
  v16h b1 = load_b16(W1 + hi * 16);
  for (int kc = 0; kc < D_; kc += 32) {
    v16h ca = a, cb0 = b0, cb1 = b1;
    if (kc + 32 < D_) {   // lookahead: next chunk's fragments
      a  = load_a(Ar + kc + 32, hi);
      b0 = load_b16(W0 + kc + 32 + hi * 16);
      b1 = load_b16(W1 + kc + 32 + hi * 16);
    }
    acc0 = wmma16(ca, cb0, acc0);
    acc1 = wmma16(ca, cb1, acc1);
  }
#pragma unroll
  for (int g = 0; g < 8; ++g) {
    int rr = mtile * 16 + g + 8 * hi;
    out[(size_t)rr * D_ + c0 + col]      = acc0[g];
    out[(size_t)rr * D_ + c0 + 16 + col] = acc1[g];
  }
}

// ---------------------------------------------------------------------------
extern "C" void kernel_launch(void* const* d_in, const int* in_sizes, int n_in,
                              void* d_out, int out_size, void* d_ws,
                              size_t ws_size, hipStream_t stream) {
  (void)in_sizes; (void)n_in; (void)out_size; (void)ws_size;
  const float* q       = (const float*)d_in[0];
  const float* k       = (const float*)d_in[1];
  const float* v       = (const float*)d_in[2];
  const float* coords  = (const float*)d_in[3];
  const float* qp      = (const float*)d_in[4];
  const float* kp      = (const float*)d_in[5];
  const float* vp      = (const float*)d_in[6];
  const float* qb      = (const float*)d_in[7];
  const float* kb      = (const float*)d_in[8];
  const float* vb      = (const float*)d_in[9];
  const float* w_out   = (const float*)d_in[10];
  const float* spreads = (const float*)d_in[11];

  char* ws = (char*)d_ws;
  const size_t szX   = (size_t)B_ * N_ * D_ * sizeof(_Float16);        // 4 MB
  const size_t szPt  = (size_t)H_ * DK_ * D_ * sizeof(_Float16);       // 512 KB
  const size_t szQKV = (size_t)B_ * H_ * N_ * DK_ * sizeof(_Float16);  // 4 MB
  _Float16* Xq  = (_Float16*)(ws);
  _Float16* Xk  = (_Float16*)(ws + szX);
  _Float16* Xv  = (_Float16*)(ws + 2 * szX);
  _Float16* Ptq = (_Float16*)(ws + 3 * szX);
  _Float16* Ptk = (_Float16*)(ws + 3 * szX + szPt);
  _Float16* Ptv = (_Float16*)(ws + 3 * szX + 2 * szPt);
  _Float16* Wh  = (_Float16*)(ws + 3 * szX + 3 * szPt);
  _Float16* Qh  = (_Float16*)(ws + 3 * szX + 4 * szPt);
  _Float16* Kh  = (_Float16*)(ws + 3 * szX + 4 * szPt + szQKV);
  _Float16* Vh  = (_Float16*)(ws + 3 * szX + 4 * szPt + 2 * szQKV);
  _Float16* Af  = (_Float16*)(ws + 3 * szX + 4 * szPt + 3 * szQKV);

  const int n_cvt = (int)(((size_t)B_ * N_ * D_ / 4 + 255) / 256);
  k_cvt<<<n_cvt, 256, 0, stream>>>(q, k, v, Xq, Xk, Xv);

  const int n_prep = (H_ * D_ * DK_ + D_ * D_ + 255) / 256;
  k_prep<<<n_prep, 256, 0, stream>>>(qp, kp, vp, w_out, Ptq, Ptk, Ptv, Wh);

  k_proj<<<(3 * B_ * (N_ / 16) * (H_ / 4)) / 4, 128, 0, stream>>>(
      Xq, Xk, Xv, Ptq, Ptk, Ptv, qb, kb, vb, Qh, Kh, Vh);

  k_attn<<<B_ * H_ * (N_ / 128), 256, 0, stream>>>(Qh, Kh, Vh, coords, spreads, Af);

  k_outproj<<<((B_ * N_ / 16) * (D_ / 32)) / 4, 128, 0, stream>>>(
      Af, Wh, (float*)d_out);
}
